// ExpandingRSSM_38955353375132
// MI455X (gfx1250) — compile-verified, matrix operand order checked by reading, not run on hardware
//
#include <hip/hip_runtime.h>
#include <hip/hip_bf16.h>
#include <stdint.h>

// Problem dims
#define B_   128
#define T_   128
#define E_   1024
#define H_   1024
#define A_   64
#define NH_  8
#define P_   512      // NH*A
#define E2_  2048     // 2E
#define H3_  3072     // 3H
#define SCALE_ 0.125f // 1/sqrt(64)

typedef __attribute__((ext_vector_type(16))) __bf16 v16bf;
typedef __attribute__((ext_vector_type(8)))  __bf16 v8bf;
typedef __attribute__((ext_vector_type(8)))  float  v8f;

__device__ __forceinline__ __bf16 f2bf(float f) { return (__bf16)f; }

// ---- CDNA5 async global->LDS path (ASYNCcnt), with sync fallback ----------
#if __has_builtin(__builtin_amdgcn_global_load_async_to_lds_b128)
#define HAVE_ASYNC_LDS 1
#else
#define HAVE_ASYNC_LDS 0
#endif

// Builtin wants pointers to 16-byte int vectors in global(1)/shared(3) AS.
typedef int v4i_ __attribute__((vector_size(16)));
typedef __attribute__((address_space(1))) v4i_* as1_v4i;
typedef __attribute__((address_space(3))) v4i_* as3_v4i;
#define AS1V(p) ((as1_v4i)(uintptr_t)(p))
#define AS3V(p) ((as3_v4i)(uintptr_t)(p))

__device__ __forceinline__ void wait_async0() {
#if __has_builtin(__builtin_amdgcn_s_wait_asynccnt)
  __builtin_amdgcn_s_wait_asynccnt(0);
#else
  asm volatile("s_wait_asynccnt 0x0" ::: "memory");
#endif
}

// ---------------------------------------------------------------------------
// Generic bf16 WMMA GEMM:  C(MxN) = A(MxK, row-major, lda) * Bt(NxK, row-major, ldb)^T
// Block: 256 threads (8 waves), tile 128(M) x 128(N), K-step 32.
// Each wave computes a 32x64 tile = 2x4 v_wmma_f32_16x16x32_bf16.
// Double-buffered LDS; tile k+1 staged with async global->LDS while tile k computes.
// epi: 0 = store f32 raw, 1 = store f32 + bias[col], 2 = store bf16 + bias[col]
// ---------------------------------------------------------------------------
__global__ __launch_bounds__(256)
void gemm_wmma_bf16(const __bf16* __restrict__ A, const __bf16* __restrict__ Bt,
                    float* __restrict__ C, __bf16* __restrict__ Cbf,
                    const float* __restrict__ bias,
                    int K, int lda, int ldb, int ldc, int epi) {
  constexpr int LS = 40;                        // LDS row stride (bf16), 80B
  __shared__ __bf16 sA[2][128 * LS];            // [buf][128 x 32]
  __shared__ __bf16 sB[2][128 * LS];

  const int tid  = threadIdx.x;
  const int lane = tid & 31;
  const int wid  = tid >> 5;
  const int wm   = (wid & 3) * 32;              // wave M offset in block tile
  const int wn   = (wid >> 2) * 64;             // wave N offset in block tile
  const size_t blockM = (size_t)blockIdx.y * 128;
  const size_t blockN = (size_t)blockIdx.x * 128;

  // staging coords: 128x32 tile = 4096 elems; 256 thr * 8 elems * 2 chunks
  const int el0 = tid * 8;
  const int r0  = el0 >> 5, c0 = el0 & 31;      // chunk 0 -> rows 0..63
  const int r1  = r0 + 64;                      // chunk 1 -> rows 64..127

  v8f acc[2][4];
#pragma unroll
  for (int mi = 0; mi < 2; ++mi)
#pragma unroll
    for (int ni = 0; ni < 4; ++ni)
#pragma unroll
      for (int t = 0; t < 8; ++t) acc[mi][ni][t] = 0.0f;

  const int nk = K >> 5;

  auto issue = [&](int kt, int buf) {
    const size_t kc = (size_t)(kt << 5) + c0;
    const __bf16* ga0 = A  + (blockM + r0) * (size_t)lda + kc;
    const __bf16* ga1 = A  + (blockM + r1) * (size_t)lda + kc;
    const __bf16* gb0 = Bt + (blockN + r0) * (size_t)ldb + kc;
    const __bf16* gb1 = Bt + (blockN + r1) * (size_t)ldb + kc;
    __bf16* la0 = &sA[buf][r0 * LS + c0];
    __bf16* la1 = &sA[buf][r1 * LS + c0];
    __bf16* lb0 = &sB[buf][r0 * LS + c0];
    __bf16* lb1 = &sB[buf][r1 * LS + c0];
#if HAVE_ASYNC_LDS
    __builtin_amdgcn_global_load_async_to_lds_b128(AS1V(ga0), AS3V(la0), 0, 0);
    __builtin_amdgcn_global_load_async_to_lds_b128(AS1V(ga1), AS3V(la1), 0, 0);
    __builtin_amdgcn_global_load_async_to_lds_b128(AS1V(gb0), AS3V(lb0), 0, 0);
    __builtin_amdgcn_global_load_async_to_lds_b128(AS1V(gb1), AS3V(lb1), 0, 0);
#else
    *(v8bf*)la0 = *(const v8bf*)ga0;
    *(v8bf*)la1 = *(const v8bf*)ga1;
    *(v8bf*)lb0 = *(const v8bf*)gb0;
    *(v8bf*)lb1 = *(const v8bf*)gb1;
    __builtin_prefetch(ga0 + 32, 0, 1);
    __builtin_prefetch(gb0 + 32, 0, 1);
#endif
  };

  issue(0, 0);

  const int fr = lane & 15;
  const int kh = (lane >> 4) * 8;

  for (int kt = 0; kt < nk; ++kt) {
    const int cur = kt & 1;
#if HAVE_ASYNC_LDS
    wait_async0();               // this wave's tile-kt async copies landed
#endif
    __syncthreads();             // everyone's tile-kt data visible / prior reads retired
    if (kt + 1 < nk) issue(kt + 1, cur ^ 1);

    // ---- fragments from LDS (16-bit A layout: lanes 0-15 -> K {0..7,16..23},
    //      lanes 16-31 -> K {8..15,24..31}; B uses same per-column layout) ----
    v16bf af[2], bfg[4];
#pragma unroll
    for (int mi = 0; mi < 2; ++mi) {
      const __bf16* p = &sA[cur][(wm + mi * 16 + fr) * LS];
#pragma unroll
      for (int t = 0; t < 8; ++t) { af[mi][t] = p[kh + t]; af[mi][8 + t] = p[16 + kh + t]; }
    }
#pragma unroll
    for (int ni = 0; ni < 4; ++ni) {
      const __bf16* p = &sB[cur][(wn + ni * 16 + fr) * LS];
#pragma unroll
      for (int t = 0; t < 8; ++t) { bfg[ni][t] = p[kh + t]; bfg[ni][8 + t] = p[16 + kh + t]; }
    }

#pragma unroll
    for (int mi = 0; mi < 2; ++mi)
#pragma unroll
      for (int ni = 0; ni < 4; ++ni)
        acc[mi][ni] = __builtin_amdgcn_wmma_f32_16x16x32_bf16(
            false, af[mi], false, bfg[ni], (short)0, acc[mi][ni], false, false);
  }

  // ---- epilogue: C/D 16x16 f32 layout: lanes 0-15 -> M=r, lanes 16-31 -> M=8+r
  const int rowOff = (lane >> 4) * 8;
  const int colOff = lane & 15;
#pragma unroll
  for (int mi = 0; mi < 2; ++mi) {
#pragma unroll
    for (int ni = 0; ni < 4; ++ni) {
      size_t rbase = blockM + wm + mi * 16 + rowOff;
      size_t cn    = blockN + wn + ni * 16 + colOff;
      float bv = (epi >= 1) ? bias[cn] : 0.0f;
#pragma unroll
      for (int r = 0; r < 8; ++r) {
        size_t off = (rbase + r) * (size_t)ldc + cn;
        float val = acc[mi][ni][r] + bv;
        if (epi == 2) Cbf[off] = f2bf(val);
        else          C[off]   = val;
      }
    }
  }
}

// ---------------------------------------------------------------------------
// Weight prep
// ---------------------------------------------------------------------------
__global__ void conv_bf16_k(__bf16* dst, const float* src, int n) {
  int i = blockIdx.x * blockDim.x + threadIdx.x;
  if (i < n) dst[i] = f2bf(src[i]);
}

// src (R x C) f32 -> dst (C x R) bf16
__global__ void transpose_bf16_k(__bf16* dst, const float* src, int R, int C) {
  int i = blockIdx.x * blockDim.x + threadIdx.x;
  if (i >= R * C) return;
  int r = i / C, c = i % C;
  dst[(size_t)c * R + r] = f2bf(src[i]);
}

__global__ void concat_bias_k(float* dst, const float* b0, const float* b1, int n) {
  int i = blockIdx.x * blockDim.x + threadIdx.x;
  if (i < n) { dst[i] = b0[i]; dst[n + i] = b1[i]; }
}

// ---------------------------------------------------------------------------
// Per-step elementwise kernels
// ---------------------------------------------------------------------------
__global__ void pack_x_k(__bf16* x_bf, const float* encs, const float* context) {
  int i = blockIdx.x * blockDim.x + threadIdx.x;
  if (i >= B_ * E2_) return;
  int b = i / E2_, c = i % E2_;
  float v = (c < E_) ? encs[(size_t)b * E_ + c] : context[(size_t)b * E_ + (c - E_)];
  x_bf[i] = f2bf(v);
}

__global__ void gru_k(const float* __restrict__ gi, const float* __restrict__ gh,
                      const float* __restrict__ b_ih, const float* __restrict__ b_hh,
                      float* __restrict__ h, __bf16* __restrict__ h_bf) {
  int i = blockIdx.x * blockDim.x + threadIdx.x;
  if (i >= B_ * H_) return;
  int b = i / H_, u = i % H_;
  size_t base = (size_t)b * H3_;
  float ir  = gi[base + u]           + b_ih[u];
  float iz  = gi[base + H_ + u]      + b_ih[H_ + u];
  float in_ = gi[base + 2 * H_ + u]  + b_ih[2 * H_ + u];
  float hr  = gh[base + u]           + b_hh[u];
  float hz  = gh[base + H_ + u]      + b_hh[H_ + u];
  float hn  = gh[base + 2 * H_ + u]  + b_hh[2 * H_ + u];
  float r = 1.0f / (1.0f + __expf(-(ir + hr)));
  float z = 1.0f / (1.0f + __expf(-(iz + hz)));
  float n = tanhf(in_ + r * hn);
  float ho = h[i];
  float hnew = (1.0f - z) * n + z * ho;
  h[i] = hnew;
  h_bf[i] = f2bf(hnew);
}

// raw (B x 2048): cols 0..1023 -> mu, 1024..2047 -> softplus(sig)
__global__ void musig_epi_k(const float* __restrict__ raw,
                            const float* __restrict__ b_mu, const float* __restrict__ b_sig,
                            float* __restrict__ mus /* = d_out */, float* __restrict__ sigs,
                            int iStep) {
  int i = blockIdx.x * blockDim.x + threadIdx.x;
  if (i >= B_ * E_) return;
  int b = i / E_, e = i % E_;
  size_t roff = (size_t)b * E2_;
  size_t off  = ((size_t)b * T_ + iStep) * E_ + e;
  float mu = raw[roff + e] + b_mu[e];
  float sp = raw[roff + E_ + e] + b_sig[e];
  float sg = (sp > 20.0f) ? sp : log1pf(__expf(sp));
  mus[off]  = mu;
  sigs[off] = sg;
}

__device__ __forceinline__ float rng_normal(uint64_t key) {
  uint64_t x = key + 0x9E3779B97F4A7C15ull;
  x ^= x >> 30; x *= 0xBF58476D1CE4E5B9ull;
  x ^= x >> 27; x *= 0x94D049BB133111EBull;
  x ^= x >> 31;
  uint32_t u1i = (uint32_t)x;
  uint32_t u2i = (uint32_t)(x >> 32);
  float u1 = ((float)u1i + 1.0f) * 2.3283064e-10f;   // (0,1]
  float u2 = (float)u2i * 2.3283064e-10f;
  float r = sqrtf(-2.0f * __logf(u1));
  return r * __cosf(6.28318531f * u2);
}

// s[b,j,:] = mu + eps(i;b,j,e) * sig for j <= iStep; also overwrite d_out on last step
__global__ void sample_k(float* __restrict__ mus /* = d_out */, const float* __restrict__ sigs,
                         __bf16* __restrict__ s_bf, int iStep, int writeOut) {
  int total = B_ * (iStep + 1) * E_;
  int i = blockIdx.x * blockDim.x + threadIdx.x;
  if (i >= total) return;
  int e  = i % E_;
  int bj = i / E_;
  int j  = bj % (iStep + 1);
  int b  = bj / (iStep + 1);
  size_t off = ((size_t)b * T_ + j) * E_ + e;
  float mu = mus[off];
  float sg = sigs[off];
  uint64_t key = (((uint64_t)(iStep + 1)) << 32) | (uint32_t)off;
  float val = mu + rng_normal(key) * sg;
  s_bf[off] = f2bf(val);
  if (writeOut) mus[off] = val;   // final output = s at step T-1
}

// Attention for query row iStep. kv (B*T x 1024) bf16: cols [0,512)=k+b_k, [512,1024)=v+b_v.
__global__ __launch_bounds__(128)
void attn_k(const float* __restrict__ q, const __bf16* __restrict__ kv,
            __bf16* __restrict__ ctx_bf, int iStep) {
  __shared__ float sc[T_];
  __shared__ float qs[A_];
  __shared__ float redmax[4], redsum[4];
  const int tid  = threadIdx.x;
  const int lane = tid & 31;
  const int wid  = tid >> 5;
  const int b  = blockIdx.x / NH_;
  const int hh = blockIdx.x % NH_;
  const float* qh = q + (size_t)b * P_ + hh * A_;
  if (tid < A_) qs[tid] = qh[tid];
  __syncthreads();

  {
    int j = tid;
    float s;
    if (j <= iStep) {
      const __bf16* kp = kv + ((size_t)b * T_ + j) * 1024 + hh * A_;
      float dot = 0.0f;
#pragma unroll 8
      for (int a = 0; a < A_; ++a) dot += qs[a] * (float)kp[a];
      s = dot * SCALE_;
    } else {
      s = -1e30f;
    }
    sc[j] = s;
  }
  float v = sc[tid];
#pragma unroll
  for (int o = 16; o; o >>= 1) v = fmaxf(v, __shfl_xor(v, o, 32));
  if (lane == 0) redmax[wid] = v;
  __syncthreads();
  float m = fmaxf(fmaxf(redmax[0], redmax[1]), fmaxf(redmax[2], redmax[3]));
  float ev = __expf(sc[tid] - m);
  float s = ev;
#pragma unroll
  for (int o = 16; o; o >>= 1) s += __shfl_xor(s, o, 32);
  if (lane == 0) redsum[wid] = s;
  __syncthreads();
  float denom = redsum[0] + redsum[1] + redsum[2] + redsum[3];
  sc[tid] = ev / denom;
  __syncthreads();

  if (tid < A_) {
    float acc = 0.0f;
    const __bf16* vp = kv + (size_t)b * T_ * 1024 + 512 + hh * A_ + tid;
    for (int j = 0; j <= iStep; ++j) acc += sc[j] * (float)vp[(size_t)j * 1024];
    ctx_bf[(size_t)b * P_ + hh * A_ + tid] = f2bf(acc);
  }
}

// ---------------------------------------------------------------------------
// Host: orchestrate 128 steps of the RSSM on `stream`
// ---------------------------------------------------------------------------
static inline size_t align256(size_t x) { return (x + 255) & ~(size_t)255; }

extern "C" void kernel_launch(void* const* d_in, const int* in_sizes, int n_in,
                              void* d_out, int out_size, void* d_ws, size_t ws_size,
                              hipStream_t stream) {
  (void)in_sizes; (void)n_in; (void)out_size; (void)ws_size;

  const float* encs  = (const float*)d_in[0];
  const float* W_ih  = (const float*)d_in[1];   // (3H, 2E)
  const float* W_hh  = (const float*)d_in[2];   // (3H, H)
  const float* b_ih  = (const float*)d_in[3];
  const float* b_hh  = (const float*)d_in[4];
  const float* W_mu  = (const float*)d_in[5];   // (H, E)
  const float* b_mu  = (const float*)d_in[6];
  const float* W_sig = (const float*)d_in[7];   // (H, E)
  const float* b_sig = (const float*)d_in[8];
  const float* W_q   = (const float*)d_in[9];   // (E, P)
  const float* b_q   = (const float*)d_in[10];
  const float* W_k   = (const float*)d_in[11];  // (E, P)
  const float* b_k   = (const float*)d_in[12];
  const float* W_v   = (const float*)d_in[13];  // (E, P)
  const float* b_v   = (const float*)d_in[14];
  const float* W_o   = (const float*)d_in[15];  // (P, E)
  const float* b_o   = (const float*)d_in[16];

  char* ws = (char*)d_ws;
  size_t o = 0;
  auto take = [&](size_t bytes) -> char* { char* p = ws + o; o += align256(bytes); return p; };

  __bf16* wih_bf   = (__bf16*)take((size_t)H3_ * E2_ * 2);      // direct (3072 x 2048)
  __bf16* whh_bf   = (__bf16*)take((size_t)H3_ * H_ * 2);       // direct (3072 x 1024)
  __bf16* wmusigT  = (__bf16*)take((size_t)E2_ * H_ * 2);       // [W_mu^T ; W_sig^T]
  __bf16* wqT      = (__bf16*)take((size_t)P_ * E_ * 2);        // (512 x 1024)
  __bf16* wkvT     = (__bf16*)take((size_t)1024 * E_ * 2);      // [W_k^T ; W_v^T]
  __bf16* woT      = (__bf16*)take((size_t)E_ * P_ * 2);        // (1024 x 512)
  float*  kvbias   = (float*)take(1024 * 4);
  float*  h        = (float*)take((size_t)B_ * H_ * 4);
  __bf16* h_bf     = (__bf16*)take((size_t)B_ * H_ * 2);
  float*  context  = (float*)take((size_t)B_ * E_ * 4);
  __bf16* x_bf     = (__bf16*)take((size_t)B_ * E2_ * 2);
  float*  gi       = (float*)take((size_t)B_ * H3_ * 4);
  float*  gh       = (float*)take((size_t)B_ * H3_ * 4);
  float*  rawms    = (float*)take((size_t)B_ * E2_ * 4);
  float*  sigs     = (float*)take((size_t)B_ * T_ * E_ * 4);
  __bf16* s_bf     = (__bf16*)take((size_t)B_ * T_ * E_ * 2);
  __bf16* kv_bf    = (__bf16*)take((size_t)B_ * T_ * 1024 * 2);
  float*  qbuf     = (float*)take((size_t)B_ * P_ * 4);
  __bf16* ctx_bf   = (__bf16*)take((size_t)B_ * P_ * 2);

  float* mus = (float*)d_out;   // (B, T, E); final step overwrites with sampled s

  const int TPB = 256;
  auto cdiv = [](int a, int b) { return (a + b - 1) / b; };

  conv_bf16_k<<<cdiv(H3_ * E2_, TPB), TPB, 0, stream>>>(wih_bf, W_ih, H3_ * E2_);
  conv_bf16_k<<<cdiv(H3_ * H_, TPB), TPB, 0, stream>>>(whh_bf, W_hh, H3_ * H_);
  transpose_bf16_k<<<cdiv(H_ * E_, TPB), TPB, 0, stream>>>(wmusigT, W_mu, H_, E_);
  transpose_bf16_k<<<cdiv(H_ * E_, TPB), TPB, 0, stream>>>(wmusigT + (size_t)E_ * H_, W_sig, H_, E_);
  transpose_bf16_k<<<cdiv(E_ * P_, TPB), TPB, 0, stream>>>(wqT, W_q, E_, P_);
  transpose_bf16_k<<<cdiv(E_ * P_, TPB), TPB, 0, stream>>>(wkvT, W_k, E_, P_);
  transpose_bf16_k<<<cdiv(E_ * P_, TPB), TPB, 0, stream>>>(wkvT + (size_t)P_ * E_, W_v, E_, P_);
  transpose_bf16_k<<<cdiv(P_ * E_, TPB), TPB, 0, stream>>>(woT, W_o, P_, E_);
  concat_bias_k<<<cdiv(P_, TPB), TPB, 0, stream>>>(kvbias, b_k, b_v, P_);

  (void)hipMemsetAsync(h,       0, (size_t)B_ * H_ * 4, stream);
  (void)hipMemsetAsync(h_bf,    0, (size_t)B_ * H_ * 2, stream);
  (void)hipMemsetAsync(context, 0, (size_t)B_ * E_ * 4, stream);
  (void)hipMemsetAsync(s_bf,    0, (size_t)B_ * T_ * E_ * 2, stream);

  for (int i = 0; i < T_; ++i) {
    pack_x_k<<<cdiv(B_ * E2_, TPB), TPB, 0, stream>>>(x_bf, encs, context);

    // gi = x @ W_ih^T : M=128, N=3072, K=2048
    gemm_wmma_bf16<<<dim3(H3_ / 128, B_ / 128), 256, 0, stream>>>(
        x_bf, wih_bf, gi, nullptr, nullptr, E2_, E2_, E2_, H3_, 0);
    // gh = h @ W_hh^T : M=128, N=3072, K=1024
    gemm_wmma_bf16<<<dim3(H3_ / 128, 1), 256, 0, stream>>>(
        h_bf, whh_bf, gh, nullptr, nullptr, H_, H_, H_, H3_, 0);

    gru_k<<<cdiv(B_ * H_, TPB), TPB, 0, stream>>>(gi, gh, b_ih, b_hh, h, h_bf);

    // [mu | sig_pre] = h @ [W_mu | W_sig] : M=128, N=2048, K=1024
    gemm_wmma_bf16<<<dim3(E2_ / 128, 1), 256, 0, stream>>>(
        h_bf, wmusigT, rawms, nullptr, nullptr, H_, H_, H_, E2_, 0);
    musig_epi_k<<<cdiv(B_ * E_, TPB), TPB, 0, stream>>>(rawms, b_mu, b_sig, mus, sigs, i);

    int tot = B_ * (i + 1) * E_;
    sample_k<<<cdiv(tot, TPB), TPB, 0, stream>>>(mus, sigs, s_bf, i, (i == T_ - 1) ? 1 : 0);

    // q_i = s[:,i] @ W_q + b_q : M=128, N=512, K=1024 (A rows strided by T*E)
    gemm_wmma_bf16<<<dim3(P_ / 128, 1), 256, 0, stream>>>(
        s_bf + (size_t)i * E_, wqT, qbuf, nullptr, b_q, E_, T_ * E_, E_, P_, 1);

    // [k|v] = s @ [W_k|W_v] + [b_k|b_v] : M=B*T=16384, N=1024, K=1024 (bf16 out)
    gemm_wmma_bf16<<<dim3(1024 / 128, (B_ * T_) / 128), 256, 0, stream>>>(
        s_bf, wkvT, nullptr, kv_bf, kvbias, E_, E_, E_, 1024, 2);

    attn_k<<<B_ * NH_, 128, 0, stream>>>(qbuf, kv_bf, ctx_bf, i);

    // context = ctx @ W_o + b_o : M=128, N=1024, K=512
    gemm_wmma_bf16<<<dim3(E_ / 128, 1), 256, 0, stream>>>(
        ctx_bf, woT, context, nullptr, b_o, P_, P_, P_, E_, 1);
  }
}
